// LocalizationModel_62045097558258
// MI455X (gfx1250) — matrix operand order repaired and checked by reading, not compiled, not run on hardware
//
#include <hip/hip_runtime.h>
#include <hip/hip_bf16.h>
#include <math.h>
#include <stdint.h>

// ---------------------------------------------------------------------------
// LocalizationModel for MI455X (gfx1250, wave32, WMMA + TDM).
// f16 NHWC activations (channel-padded), implicit-GEMM conv via
// v_wmma_f32_16x16x32_f16; halo windows staged into LDS by the Tensor Data
// Mover (tensor_load_to_lds, TENSORcnt) when available.
// ---------------------------------------------------------------------------

typedef __attribute__((ext_vector_type(16))) _Float16 v16h;
typedef __attribute__((ext_vector_type(8)))  _Float16 h8;
typedef __attribute__((ext_vector_type(8)))  float    v8f;
typedef unsigned int u32x4 __attribute__((ext_vector_type(4)));
typedef int          i32x4 __attribute__((ext_vector_type(4)));
typedef int          i32x8 __attribute__((ext_vector_type(8)));

#define S_ 16384   // 128*128
#define H_ 128
#define W_ 128
#define B_ 4

#if defined(__has_builtin)
#if __has_builtin(__builtin_amdgcn_tensor_load_to_lds)
#define HAVE_TDM 1
#endif
#if __has_builtin(__builtin_amdgcn_s_wait_tensorcnt)
#define HAVE_TENSORCNT 1
#endif
#endif

#if defined(HAVE_TDM)
typedef __attribute__((address_space(3))) const void* lds_cptr;
__device__ __forceinline__ unsigned lds_off(const void* p) {
  return (unsigned)(uintptr_t)(lds_cptr)p;
}
// One TDM descriptor: copy `elems` f16 elements (1 row) from global to LDS.
// D# group0: count=1 | lds_addr | global_addr | type=2.
// D# group1: data_size=2B; tensor_dim0=tile_dim0=elems; tensor_dim1=tile_dim1=1;
//            tensor_dim0_stride=elems; no padding/iterate/multicast.
__device__ __forceinline__ void tdm_load_row(unsigned lds_byte,
                                             unsigned long long gaddr,
                                             unsigned elems) {
  u32x4 g0;
  g0.x = 1u;                                      // count=1, user descriptor
  g0.y = lds_byte;                                // LDS dest (bytes)
  g0.z = (unsigned)gaddr;                         // global_addr[31:0]
  g0.w = (unsigned)(gaddr >> 32) | 0x80000000u;   // global_addr[56:32] | type=2
  i32x8 g1 = {};
  g1[0] = 1 << 16;                                // data_size = 2 bytes
  g1[1] = (int)((elems & 0xFFFFu) << 16);         // tensor_dim0[15:0]
  g1[2] = (int)((elems >> 16) | (1u << 16));      // tensor_dim0[31:16] | dim1=1
  g1[3] = (int)((elems & 0xFFFFu) << 16);         // tile_dim0 = elems
  g1[4] = 1;                                      // tile_dim1 = 1
  g1[5] = (int)elems;                             // tensor_dim0_stride
  i32x4 z4 = {};
#if __clang_major__ >= 23
  i32x8 z8 = {};
  __builtin_amdgcn_tensor_load_to_lds(g0, g1, z4, z4, z8, 0);
#else
  __builtin_amdgcn_tensor_load_to_lds(g0, g1, z4, z4, 0);
#endif
}
#endif

// ---------------------------------------------------------------------------
// f32 NCHW (C=4/3) -> f16 NHWC with Cpad=32 (zero padded channels)
// ---------------------------------------------------------------------------
__global__ __launch_bounds__(256) void to_nhwc32(const float* __restrict__ in,
                                                 _Float16* __restrict__ outp,
                                                 int C) {
  const size_t idx = (size_t)blockIdx.x * 256 + threadIdx.x;  // B*S*32
  const int c = idx & 31;
  const int p = (idx >> 5) & 16383;
  const int b = idx >> 19;
  float v = (c < C) ? in[((size_t)(b * C + c) << 14) + p] : 0.f;
  outp[idx] = (_Float16)v;
}

// ---------------------------------------------------------------------------
// Pack weights transposed: w[(Cout,Cin,3,3)] f32 -> wpT[n][j*Cinp + c] f16.
// ---------------------------------------------------------------------------
__global__ __launch_bounds__(256) void pack_wT(const float* __restrict__ w,
                                               _Float16* __restrict__ wp,
                                               int Cin, int Cinp, int Cout,
                                               int taps) {
  const int total = Cout * taps * Cinp;
  const int idx = blockIdx.x * 256 + threadIdx.x;
  if (idx >= total) return;
  const int c = idx % Cinp;
  const int j = (idx / Cinp) % taps;
  const int n = idx / (Cinp * taps);
  float v = (c < Cin) ? w[((size_t)n * Cin + c) * taps + j] : 0.f;
  wp[idx] = (_Float16)v;
}

// ---------------------------------------------------------------------------
// conv3x3 implicit GEMM. Block: (b, y, 64-px half row) x 16 Cout.
// LDS tile: 3 rows x 66 px x CINP halfs, pre-zeroed; valid 65-px runs are
// DMA'd by the TDM (or b128-copied on toolchains without the builtin).
// ---------------------------------------------------------------------------
template <int CINP>
__global__ __launch_bounds__(128) void conv3x3_wmma(
    const _Float16* __restrict__ act,   // f16 NHWC [B][128][128][CINP]
    const _Float16* __restrict__ wpT,   // f16 [Cout][9*CINP]
    float* __restrict__ out,            // f32 NCHW [B][Cout][128][128]
    int Cout) {
  constexpr int KPAD = 9 * CINP;
  __shared__ _Float16 sIn[3 * 66 * CINP];

  const int tid  = threadIdx.x;
  const int lane = tid & 31, wave = tid >> 5;
  const int grp  = lane >> 4, lr = lane & 15;

  const int bx = blockIdx.x;           // B*128*2
  const int xh = bx & 1;
  const int y  = (bx >> 1) & 127;
  const int b  = bx >> 8;
  const int x0 = xh * 64;
  const int nbase = blockIdx.y * 16;

  // ---- pre-zero the tile (provides all halo zeros, branch-free K loop) ----
  constexpr int TOT8 = 3 * 66 * CINP / 8;
  {
    const h8 zz = {};
    for (int u = tid; u < TOT8; u += 128) *(h8*)(sIn + (size_t)u * 8) = zz;
  }
  __syncthreads();

  const int pxv = xh ? 0 : 1;          // first valid px slot in the 66-px row
  const int gx0 = xh ? 63 : 0;         // first valid global x
#if defined(HAVE_TDM)
  if (tid < 32) {                      // wave 0 only: TDM ignores EXEC
#pragma unroll
    for (int r = 0; r < 3; ++r) {
      const int gy = y + r - 1;
      if (gy >= 0 && gy < H_) {
        const unsigned ldsb =
            lds_off(sIn + ((size_t)(r * 66) + pxv) * CINP);
        const unsigned long long ga = (unsigned long long)(uintptr_t)(
            act + ((size_t)(b * H_ + gy) * W_ + gx0) * CINP);
        tdm_load_row(ldsb, ga, 65u * CINP);
      }
    }
#if defined(HAVE_TENSORCNT)
    __builtin_amdgcn_s_wait_tensorcnt(0);
#endif
  }
#else
  {
    constexpr int RUN8 = 65 * CINP / 8;
    for (int u = tid; u < 3 * RUN8; u += 128) {
      const int r = u / RUN8;
      const int o = (u - r * RUN8) * 8;
      const int gy = y + r - 1;
      if (gy >= 0 && gy < H_) {
        const h8 v =
            *(const h8*)(act + (((size_t)(b * H_ + gy) * W_ + gx0) * CINP + o));
        *(h8*)(sIn + ((size_t)(r * 66) + pxv) * CINP + o) = v;
      }
    }
  }
#endif
  __syncthreads();

  const int px = wave * 16 + lr;                    // A-row pixel (0..63)
  const _Float16* wrow = wpT + (size_t)(nbase + lr) * KPAD;

  v8f acc = {};
#pragma unroll
  for (int j = 0; j < 9; ++j) {
    const int r = j / 3, dxx = j % 3;
    const _Float16* ain = sIn + ((size_t)(r * 66) + px + dxx) * CINP;
    const _Float16* brow = wrow + j * CINP;
#pragma unroll
    for (int c0 = 0; c0 < CINP; c0 += 32) {
      const h8 alo = *(const h8*)(ain + c0 + grp * 8);
      const h8 ahi = *(const h8*)(ain + c0 + 16 + grp * 8);
      const h8 blo = *(const h8*)(brow + c0 + grp * 8);
      const h8 bhi = *(const h8*)(brow + c0 + 16 + grp * 8);
      const v16h a  = __builtin_shufflevector(alo, ahi, 0, 1, 2, 3, 4, 5, 6, 7,
                                              8, 9, 10, 11, 12, 13, 14, 15);
      const v16h bf = __builtin_shufflevector(blo, bhi, 0, 1, 2, 3, 4, 5, 6, 7,
                                              8, 9, 10, 11, 12, 13, 14, 15);
      acc = __builtin_amdgcn_wmma_f32_16x16x32_f16(false, a, false, bf,
                                                   (short)0, acc, false, false);
    }
  }

  const int n = nbase + lr;
  float* ob = out + (((size_t)b * Cout + n) << 14) + (y << 7);
#pragma unroll
  for (int v = 0; v < 8; ++v)
    ob[x0 + wave * 16 + v + grp * 8] = acc[v];
}

// ---------------------------------------------------------------------------
// 1x1 projection: K = 128 channels, contiguous fragments straight from global.
// ---------------------------------------------------------------------------
__global__ __launch_bounds__(128) void conv1x1_wmma(
    const _Float16* __restrict__ act,   // f16 NHWC [B*S][128]
    const _Float16* __restrict__ wpT,   // f16 [64][128]
    float* __restrict__ out) {          // f32 NCHW [B][64][S]
  const int lane = threadIdx.x & 31, wave = threadIdx.x >> 5;
  const int grp = lane >> 4, lr = lane & 15;
  const int mbase = (blockIdx.x * 4 + wave) * 16;
  const int nbase = blockIdx.y * 16;
  const int m = mbase + lr;
  const _Float16* ain  = act + (size_t)m * 128;
  const _Float16* wrow = wpT + (size_t)(nbase + lr) * 128;

  v8f acc = {};
#pragma unroll
  for (int c0 = 0; c0 < 128; c0 += 32) {
    const h8 alo = *(const h8*)(ain + c0 + grp * 8);
    const h8 ahi = *(const h8*)(ain + c0 + 16 + grp * 8);
    const h8 blo = *(const h8*)(wrow + c0 + grp * 8);
    const h8 bhi = *(const h8*)(wrow + c0 + 16 + grp * 8);
    const v16h a  = __builtin_shufflevector(alo, ahi, 0, 1, 2, 3, 4, 5, 6, 7,
                                            8, 9, 10, 11, 12, 13, 14, 15);
    const v16h bf = __builtin_shufflevector(blo, bhi, 0, 1, 2, 3, 4, 5, 6, 7,
                                            8, 9, 10, 11, 12, 13, 14, 15);
    acc = __builtin_amdgcn_wmma_f32_16x16x32_f16(false, a, false, bf,
                                                 (short)0, acc, false, false);
  }

  const int n = nbase + lr;
#pragma unroll
  for (int v = 0; v < 8; ++v) {
    const int mm = mbase + v + grp * 8;
    const int b = mm >> 14, p = mm & 16383;
    out[(((size_t)b * 64 + n) << 14) + p] = acc[v];
  }
}

// ---------------------------------------------------------------------------
// Instance norm + affine + leaky ReLU: f32 NCHW in -> f16 NHWC out.
// ---------------------------------------------------------------------------
__global__ __launch_bounds__(256) void inorm_lrelu(
    const float* __restrict__ in, const float* __restrict__ g,
    const float* __restrict__ bt, _Float16* __restrict__ out, int C) {
  const int bc = blockIdx.x;
  const int b = bc / C, c = bc % C;
  const float* src = in + (size_t)bc * S_;
  __shared__ float r1[256], r2[256];
  float s = 0.f, s2 = 0.f;
  for (int i = threadIdx.x; i < S_; i += 256) {
    float v = src[i];
    s += v; s2 += v * v;
  }
  r1[threadIdx.x] = s; r2[threadIdx.x] = s2;
  __syncthreads();
  for (int o = 128; o > 0; o >>= 1) {
    if (threadIdx.x < o) { r1[threadIdx.x] += r1[threadIdx.x + o];
                           r2[threadIdx.x] += r2[threadIdx.x + o]; }
    __syncthreads();
  }
  const float mu  = r1[0] * (1.f / S_);
  const float var = r2[0] * (1.f / S_) - mu * mu;
  const float rs  = rsqrtf(var + 1e-5f);
  const float gg = g[c], bb = bt[c];
  _Float16* dst = out + (size_t)b * S_ * C + c;
  for (int i = threadIdx.x; i < S_; i += 256) {
    float v = (src[i] - mu) * rs * gg + bb;
    v = (v >= 0.f) ? v : 0.1f * v;
    dst[(size_t)i * C] = (_Float16)v;
  }
}

// channel mean over f16 NHWC (Cpad=64). One block per (b,c).
__global__ __launch_bounds__(256) void channel_mean(
    const _Float16* __restrict__ act, float* __restrict__ outm) {
  const int bc = blockIdx.x;               // b*64 + c
  const int b = bc >> 6, c = bc & 63;
  const _Float16* src = act + (size_t)b * S_ * 64 + c;
  __shared__ float red[256];
  float s = 0.f;
  for (int i = threadIdx.x; i < S_; i += 256) s += (float)src[(size_t)i * 64];
  red[threadIdx.x] = s;
  __syncthreads();
  for (int o = 128; o > 0; o >>= 1) {
    if (threadIdx.x < o) red[threadIdx.x] += red[threadIdx.x + o];
    __syncthreads();
  }
  if (threadIdx.x == 0) outm[bc] = red[0] * (1.f / S_);
}

// attention MLP: 64 -> relu(32) -> sigmoid(64). One block per batch.
__global__ __launch_bounds__(64) void attn_mlp(
    const float* __restrict__ ctx, const float* __restrict__ w1,
    const float* __restrict__ b1, const float* __restrict__ w2,
    const float* __restrict__ b2, float* __restrict__ aw) {
  __shared__ float h[32];
  const int b = blockIdx.x, t = threadIdx.x;
  if (t < 32) {
    float s = b1[t];
    for (int c = 0; c < 64; ++c) s += w1[t * 64 + c] * ctx[b * 64 + c];
    h[t] = s > 0.f ? s : 0.f;
  }
  __syncthreads();
  if (t < 64) {
    float s = b2[t];
    for (int i = 0; i < 32; ++i) s += w2[t * 32 + i] * h[i];
    aw[b * 64 + t] = 1.f / (1.f + expf(-s));
  }
}

// per-channel scale of f16 NHWC (Cpad=64), in place.
__global__ __launch_bounds__(256) void scale16(_Float16* __restrict__ x,
                                               const float* __restrict__ aw) {
  const size_t i = (size_t)blockIdx.x * 256 + threadIdx.x;  // B*S*64
  const int c = i & 63;
  const int b = i >> 20;
  x[i] = (_Float16)((float)x[i] * aw[b * 64 + c]);
}

// L2 normalize over 64 channels per pixel + nan_to_num (f32 NCHW -> f32 NCHW).
__global__ __launch_bounds__(256) void l2norm_nan(const float* __restrict__ raw,
                                                  float* __restrict__ outp) {
  const int idx = blockIdx.x * 256 + threadIdx.x;  // < B*S
  const int b = idx >> 14, p = idx & 16383;
  const float* src = raw + ((size_t)b * 64 << 14);
  float ss = 0.f;
  for (int c = 0; c < 64; ++c) {
    float v = src[((size_t)c << 14) + p];
    ss += v * v;
  }
  const float inv = 1.f / fmaxf(sqrtf(ss), 1e-6f);
  for (int c = 0; c < 64; ++c) {
    float v = src[((size_t)c << 14) + p] * inv;
    if (isnan(v)) v = 0.f;
    if (isinf(v)) v = (v > 0.f) ? 1e6f : -1e6f;
    outp[((size_t)(b * 64 + c) << 14) + p] = v;
  }
}

// f32 NCHW (64ch) -> f16 pixel-major [b][y][x][64] for tcost fragments.
__global__ __launch_bounds__(256) void nchw_to_pix64(
    const float* __restrict__ in, _Float16* __restrict__ outp) {
  const size_t idx = (size_t)blockIdx.x * 256 + threadIdx.x;  // B*S*64
  const int c = idx & 63;
  const int p = (idx >> 6) & 16383;
  const int b = idx >> 20;
  outp[idx] = (_Float16)in[((size_t)(b * 64 + c) << 14) + p];
}

__global__ __launch_bounds__(256) void zero_f(float* __restrict__ p, int n) {
  int i = blockIdx.x * 256 + threadIdx.x;
  if (i < n) p[i] = 0.f;
}

// ---------------------------------------------------------------------------
// tcost via WMMA, K ordered as x*64 + c (contiguous channels):
//   tcost[dy,dx] += sum_{c,x} lf[c, Y-dy, x] * gp[c, Y, x+dx]  per gp row Y.
// 48x48 padded tile (3x3 waves), accumulated over Y, atomicAdd into d_out.
// ---------------------------------------------------------------------------
#define YPER 8
__global__ __launch_bounds__(288) void tcost_wmma(
    const _Float16* __restrict__ lfh, const _Float16* __restrict__ gfh,
    float* __restrict__ tc) {
  const int lane = threadIdx.x & 31;
  const int wave = threadIdx.x >> 5;     // 0..8
  const int grp = lane >> 4, lr = lane & 15;
  const int ti = wave / 3, tj = wave % 3;
  const int dyb = ti * 16, dxb = tj * 16;
  const int b = blockIdx.x;
  const int y0c = blockIdx.y * YPER;
  const int dyA = dyb + lr;   // this lane's A row (dy)
  const int dxB = dxb + lr;   // this lane's B col (dx)
  const _Float16* lfb = lfh + (size_t)b * S_ * 64;
  const _Float16* gfb = gfh + (size_t)b * S_ * 64;

  v8f acc = {};
  for (int Y = y0c; Y < y0c + YPER; ++Y) {
    const int  ylf  = Y - dyA;
    const bool avld = (ylf >= 0) && (ylf < H_);
    const int  ygf  = Y - 16;
    const bool bvld = (ygf >= 0) && (ygf < H_);
    const _Float16* arow = lfb + ((size_t)(avld ? ylf : 0) << 7) * 64;
    const _Float16* brow = gfb + ((size_t)(bvld ? ygf : 0) << 7) * 64;
    for (int x = 0; x < W_; ++x) {
      const int X = x + dxB - 16;
      const bool bok = bvld && (X >= 0) && (X < W_);
      const _Float16* ap = arow + (size_t)x * 64;
      const _Float16* bp = brow + (size_t)(bok ? X : 0) * 64;
#pragma unroll
      for (int c0 = 0; c0 < 64; c0 += 32) {
        h8 alo = {}, ahi = {}, blo = {}, bhi = {};
        if (avld) {
          alo = *(const h8*)(ap + c0 + grp * 8);
          ahi = *(const h8*)(ap + c0 + 16 + grp * 8);
        }
        if (bok) {
          blo = *(const h8*)(bp + c0 + grp * 8);
          bhi = *(const h8*)(bp + c0 + 16 + grp * 8);
        }
        const v16h a  = __builtin_shufflevector(alo, ahi, 0, 1, 2, 3, 4, 5, 6,
                                                7, 8, 9, 10, 11, 12, 13, 14, 15);
        const v16h bf = __builtin_shufflevector(blo, bhi, 0, 1, 2, 3, 4, 5, 6,
                                                7, 8, 9, 10, 11, 12, 13, 14, 15);
        acc = __builtin_amdgcn_wmma_f32_16x16x32_f16(false, a, false, bf,
                                                     (short)0, acc, false, false);
      }
    }
  }
#pragma unroll
  for (int v = 0; v < 8; ++v) {
    const int dy = dyb + v + (grp << 3);
    const int dx = dxb + lr;
    if (dy < 33 && dx < 33)
      atomicAdd(&tc[b * 1089 + dy * 33 + dx], acc[v]);
  }
}

// ---------------------------------------------------------------------------
// ocost: bilinear-rotated lf dotted with gf, 11 angles. Block per (b, theta).
// ---------------------------------------------------------------------------
__device__ __forceinline__ float gat(const float* base, int yy, int xx) {
  int yc = yy < 0 ? 0 : (yy > 127 ? 127 : yy);
  int xc = xx < 0 ? 0 : (xx > 127 ? 127 : xx);
  float v = base[(yc << 7) + xc];
  return (yy >= 0 && yy < 128 && xx >= 0 && xx < 128) ? v : 0.f;
}

__global__ __launch_bounds__(256) void ocost_kernel(
    const float* __restrict__ lf, const float* __restrict__ gf,
    float* __restrict__ oc) {
  const int b = blockIdx.x, kb = blockIdx.y;
  const float th = (-10.f + 2.f * kb) * 0.017453292519943295f;
  const float ct = cosf(th), st = sinf(th);
  const float* lfb = lf + (size_t)b * 64 * S_;
  const float* gfb = gf + (size_t)b * 64 * S_;
  float part = 0.f;
  for (int p = threadIdx.x; p < S_; p += 256) {
    const int y = p >> 7, x = p & 127;
    const float xs = (2.f * x + 1.f) * (1.f / 128.f) - 1.f;
    const float ys = (2.f * y + 1.f) * (1.f / 128.f) - 1.f;
    const float gx = ct * xs - st * ys;
    const float gy = st * xs + ct * ys;
    const float ix = ((gx + 1.f) * 128.f - 1.f) * 0.5f;
    const float iy = ((gy + 1.f) * 128.f - 1.f) * 0.5f;
    const int x0 = (int)floorf(ix), y0 = (int)floorf(iy);
    const float wx = ix - x0, wy = iy - y0;
    const float w00 = (1.f - wy) * (1.f - wx), w01 = (1.f - wy) * wx;
    const float w10 = wy * (1.f - wx),         w11 = wy * wx;
    for (int c = 0; c < 64; ++c) {
      const float* pl = lfb + ((size_t)c << 14);
      float rot = gat(pl, y0, x0) * w00 + gat(pl, y0, x0 + 1) * w01 +
                  gat(pl, y0 + 1, x0) * w10 + gat(pl, y0 + 1, x0 + 1) * w11;
      part += rot * gfb[((size_t)c << 14) + p];
    }
  }
  __shared__ float red[256];
  red[threadIdx.x] = part;
  __syncthreads();
  for (int o = 128; o > 0; o >>= 1) {
    if (threadIdx.x < o) red[threadIdx.x] += red[threadIdx.x + o];
    __syncthreads();
  }
  if (threadIdx.x == 0) oc[b * 11 + kb] = red[0];
}

// ---------------------------------------------------------------------------
// Orchestration
// ---------------------------------------------------------------------------
extern "C" void kernel_launch(void* const* d_in, const int* in_sizes, int n_in,
                              void* d_out, int out_size, void* d_ws, size_t ws_size,
                              hipStream_t stream) {
  (void)in_sizes; (void)n_in; (void)out_size; (void)ws_size;

  const float* lidar = (const float*)d_in[0];
  const float* geos  = (const float*)d_in[1];
  auto PP = [&](int i) { return (const float*)d_in[i]; };
  const float *on_w[4], *on_g[4], *on_b[4], *ge_w[4], *ge_g[4], *ge_b[4];
  for (int i = 0; i < 4; ++i) {
    on_w[i] = PP(2 + 6 * i); on_g[i] = PP(3 + 6 * i); on_b[i] = PP(4 + 6 * i);
    ge_w[i] = PP(5 + 6 * i); ge_g[i] = PP(6 + 6 * i); ge_b[i] = PP(7 + 6 * i);
  }
  const float* aw1 = PP(26); const float* ab1 = PP(27);
  const float* aw2 = PP(28); const float* ab2 = PP(29);
  const float* pon = PP(30); const float* pge = PP(31);

  // ---- workspace carving ---------------------------------------------------
  _Float16* actA = (_Float16*)d_ws;                 // B*S*128 halfs (16 MB)
  _Float16* actB = actA + (size_t)8388608;          // 16 MB
  float* bufF = (float*)(actB + (size_t)8388608);   // f32 NCHW scratch, 32 MB
  _Float16* lfh = (_Float16*)(bufF + (size_t)8388608);  // 8 MB
  _Float16* gfh = lfh + (size_t)4194304;            // 8 MB
  char* cur = (char*)(gfh + (size_t)4194304);
  auto carveH = [&](size_t n) { _Float16* p = (_Float16*)cur; cur += n * 2; return p; };
  const int onCin[4] = {4, 64, 128, 128}, onCinp[4] = {32, 64, 128, 128},
            onCout[4] = {64, 128, 128, 128};
  const int geCin[4] = {3, 64, 128, 128}, geCinp[4] = {32, 64, 128, 128},
            geCout[4] = {64, 128, 128, 128};
  _Float16* wpOn[4]; _Float16* wpGe[4];
  for (int i = 0; i < 4; ++i) wpOn[i] = carveH((size_t)9 * onCinp[i] * onCout[i]);
  for (int i = 0; i < 4; ++i) wpGe[i] = carveH((size_t)9 * geCinp[i] * geCout[i]);
  _Float16* wpPon = carveH(64 * 128);
  _Float16* wpPge = carveH(64 * 128);
  cur = (char*)(((uintptr_t)cur + 15) & ~(uintptr_t)15);
  float* ctx = (float*)cur; cur += 256 * sizeof(float);
  float* awv = (float*)cur; cur += 256 * sizeof(float);

  // ---- weight packing ------------------------------------------------------
  auto pack = [&](const float* w, _Float16* wp, int Cin, int Cinp, int Cout,
                  int taps) {
    int n = Cout * taps * Cinp;
    pack_wT<<<(n + 255) / 256, 256, 0, stream>>>(w, wp, Cin, Cinp, Cout, taps);
  };
  for (int i = 0; i < 4; ++i) {
    pack(on_w[i], wpOn[i], onCin[i], onCinp[i], onCout[i], 9);
    pack(ge_w[i], wpGe[i], geCin[i], geCinp[i], geCout[i], 9);
  }
  pack(pon, wpPon, 128, 128, 64, 1);
  pack(pge, wpPge, 128, 128, 64, 1);

  float* lfp = (float*)d_out;
  float* gfp = lfp + (size_t)4194304;
  float* tcp = lfp + (size_t)8388608;
  float* ocp = lfp + (size_t)8392964;

  // ---- ON branch -----------------------------------------------------------
  to_nhwc32<<<8192, 256, 0, stream>>>(lidar, actA, 4);
  conv3x3_wmma<32><<<dim3(1024, 4), 128, 0, stream>>>(actA, wpOn[0], bufF, 64);
  inorm_lrelu<<<B_ * 64, 256, 0, stream>>>(bufF, on_g[0], on_b[0], actB, 64);
  channel_mean<<<B_ * 64, 256, 0, stream>>>(actB, ctx);
  attn_mlp<<<B_, 64, 0, stream>>>(ctx, aw1, ab1, aw2, ab2, awv);
  scale16<<<16384, 256, 0, stream>>>(actB, awv);
  conv3x3_wmma<64><<<dim3(1024, 8), 128, 0, stream>>>(actB, wpOn[1], bufF, 128);
  inorm_lrelu<<<B_ * 128, 256, 0, stream>>>(bufF, on_g[1], on_b[1], actA, 128);
  conv3x3_wmma<128><<<dim3(1024, 8), 128, 0, stream>>>(actA, wpOn[2], bufF, 128);
  inorm_lrelu<<<B_ * 128, 256, 0, stream>>>(bufF, on_g[2], on_b[2], actB, 128);
  conv3x3_wmma<128><<<dim3(1024, 8), 128, 0, stream>>>(actB, wpOn[3], bufF, 128);
  inorm_lrelu<<<B_ * 128, 256, 0, stream>>>(bufF, on_g[3], on_b[3], actA, 128);
  conv1x1_wmma<<<dim3(1024, 4), 128, 0, stream>>>(actA, wpPon, bufF);
  l2norm_nan<<<256, 256, 0, stream>>>(bufF, lfp);

  // ---- GEO branch ----------------------------------------------------------
  to_nhwc32<<<8192, 256, 0, stream>>>(geos, actA, 3);
  conv3x3_wmma<32><<<dim3(1024, 4), 128, 0, stream>>>(actA, wpGe[0], bufF, 64);
  inorm_lrelu<<<B_ * 64, 256, 0, stream>>>(bufF, ge_g[0], ge_b[0], actB, 64);
  conv3x3_wmma<64><<<dim3(1024, 8), 128, 0, stream>>>(actB, wpGe[1], bufF, 128);
  inorm_lrelu<<<B_ * 128, 256, 0, stream>>>(bufF, ge_g[1], ge_b[1], actA, 128);
  conv3x3_wmma<128><<<dim3(1024, 8), 128, 0, stream>>>(actA, wpGe[2], bufF, 128);
  inorm_lrelu<<<B_ * 128, 256, 0, stream>>>(bufF, ge_g[2], ge_b[2], actB, 128);
  conv3x3_wmma<128><<<dim3(1024, 8), 128, 0, stream>>>(actB, wpGe[3], bufF, 128);
  inorm_lrelu<<<B_ * 128, 256, 0, stream>>>(bufF, ge_g[3], ge_b[3], actA, 128);
  conv1x1_wmma<<<dim3(1024, 4), 128, 0, stream>>>(actA, wpPge, bufF);
  l2norm_nan<<<256, 256, 0, stream>>>(bufF, gfp);

  // ---- costs ---------------------------------------------------------------
  nchw_to_pix64<<<16384, 256, 0, stream>>>(lfp, lfh);
  nchw_to_pix64<<<16384, 256, 0, stream>>>(gfp, gfh);
  zero_f<<<(4356 + 255) / 256, 256, 0, stream>>>(tcp, 4356);
  tcost_wmma<<<dim3(B_, 160 / YPER), 288, 0, stream>>>(lfh, gfh, tcp);
  ocost_kernel<<<dim3(B_, 11), 256, 0, stream>>>(lfp, gfp, ocp);
}
